// DreamGMFlow_89670327205994
// MI455X (gfx1250) — matrix operand order repaired and checked by reading, not compiled, not run on hardware
//
#include <hip/hip_runtime.h>
#include <hip/hip_bf16.h>
#include <math.h>

// ---------------- problem constants ----------------
#define BB   256
#define SS   50
#define DD   128
#define KK   4
#define NN   50000
#define HH   2
#define DHH  64
#define DFFF 2048
#define NLL  2

typedef unsigned short u16;
typedef __attribute__((ext_vector_type(8)))  u16     v8u;
typedef __attribute__((ext_vector_type(16))) u16     v16u;
typedef __attribute__((ext_vector_type(16))) __bf16  v16bf;
typedef __attribute__((ext_vector_type(8)))  float   v8f;

// ---------------- helpers ----------------
__device__ __forceinline__ u16 f2bf(float x) {
  unsigned u = __builtin_bit_cast(unsigned, x);
  unsigned r = u + 0x7FFFu + ((u >> 16) & 1u);   // round-to-nearest-even
  return (u16)(r >> 16);
}
__device__ __forceinline__ float wsum(float v) {   // wave32 reduction
  for (int o = 16; o; o >>= 1) v += __shfl_xor(v, o, 32);
  return v;
}
__device__ __forceinline__ float silu_f(float x) { return x / (1.0f + __expf(-x)); }

// Load A fragment (16x32 bf16, M=lane%16; K 0-15 vs 16-31 per lane half)
__device__ __forceinline__ v16bf load_a(const u16* __restrict__ row, int k0, int half) {
  v16u av = *(const v16u*)(row + k0 + half * 16);
  return __builtin_bit_cast(v16bf, av);
}
// Load B fragment (32x16 bf16) from a row-major [N,K] matrix: N=lane%16, K half per lane half
__device__ __forceinline__ v16bf load_b(const u16* __restrict__ row, int k0, int half) {
  v16u bv = *(const v16u*)(row + k0 + half * 16);
  return __builtin_bit_cast(v16bf, bv);
}

// ---------------- kernel 1: l2norm item table -> bf16 + nsq ----------------
__global__ void k_items(const float* __restrict__ item_emb, u16* __restrict__ items_bf,
                        float* __restrict__ nsq) {
  int row = blockIdx.x * 8 + (threadIdx.x >> 5);
  if (row >= NN) return;
  int lane = threadIdx.x & 31;
  float4 v = *(const float4*)(item_emb + (size_t)row * DD + lane * 4);
  float ss = wsum(v.x*v.x + v.y*v.y + v.z*v.z + v.w*v.w);
  float d = fmaxf(sqrtf(ss), 1e-12f);
  float inv = 1.0f / d;
  u16* op = items_bf + (size_t)row * DD + lane * 4;
  op[0] = f2bf(v.x*inv); op[1] = f2bf(v.y*inv); op[2] = f2bf(v.z*inv); op[3] = f2bf(v.w*inv);
  if (lane == 0) nsq[row] = ss * inv * inv;
}

// ---------------- kernel 2: f32 -> bf16 convert ----------------
__global__ void k_cvt(const float* __restrict__ src, u16* __restrict__ dst, int n) {
  int i = blockIdx.x * 256 + threadIdx.x;
  if (i < n) dst[i] = f2bf(src[i]);
}

// ---------------- kernel 3: embedding + l2norm + pos + LN0 ----------------
__global__ void k_embed(const float* __restrict__ item_emb, const float* __restrict__ pos_emb,
                        const int* __restrict__ seq, const float* __restrict__ lw,
                        const float* __restrict__ lb, float* __restrict__ x, u16* __restrict__ xb) {
  int row = blockIdx.x * 8 + (threadIdx.x >> 5);
  if (row >= BB * SS) return;
  int lane = threadIdx.x & 31;
  int s = row % SS;
  int idx = seq[row];
  float4 e = *(const float4*)(item_emb + (size_t)idx * DD + lane * 4);
  float ss = wsum(e.x*e.x + e.y*e.y + e.z*e.z + e.w*e.w);
  float inv = 1.0f / fmaxf(sqrtf(ss), 1e-12f);
  float4 p = *(const float4*)(pos_emb + (size_t)s * DD + lane * 4);
  float y[4] = { e.x*inv + p.x, e.y*inv + p.y, e.z*inv + p.z, e.w*inv + p.w };
  float m = wsum(y[0] + y[1] + y[2] + y[3]) * (1.0f / DD);
  float d0 = y[0]-m, d1 = y[1]-m, d2 = y[2]-m, d3 = y[3]-m;
  float var = wsum(d0*d0 + d1*d1 + d2*d2 + d3*d3) * (1.0f / DD);
  float r = rsqrtf(var + 1e-5f);
  float* xo = x + (size_t)row * DD + lane * 4;
  u16*   bo = xb + (size_t)row * DD + lane * 4;
#pragma unroll
  for (int j = 0; j < 4; ++j) {
    float o = (y[j] - m) * r * lw[lane*4+j] + lb[lane*4+j];
    xo[j] = o; bo[j] = f2bf(o);
  }
}

// ---------------- kernel 4: residual + LN (in-place on x, also bf16 copy) ----------------
__global__ void k_resln(float* __restrict__ x, const float* __restrict__ y,
                        const float* __restrict__ lw, const float* __restrict__ lb,
                        u16* __restrict__ xb) {
  int row = blockIdx.x * 8 + (threadIdx.x >> 5);
  if (row >= BB * SS) return;
  int lane = threadIdx.x & 31;
  float* xr = x + (size_t)row * DD + lane * 4;
  const float* yr = y + (size_t)row * DD + lane * 4;
  float v[4];
#pragma unroll
  for (int j = 0; j < 4; ++j) v[j] = xr[j] + yr[j];
  float m = wsum(v[0]+v[1]+v[2]+v[3]) * (1.0f / DD);
  float d0=v[0]-m, d1=v[1]-m, d2=v[2]-m, d3=v[3]-m;
  float var = wsum(d0*d0+d1*d1+d2*d2+d3*d3) * (1.0f / DD);
  float r = rsqrtf(var + 1e-5f);
  u16* bo = xb + (size_t)row * DD + lane * 4;
#pragma unroll
  for (int j = 0; j < 4; ++j) {
    float o = (v[j]-m) * r * lw[lane*4+j] + lb[lane*4+j];
    xr[j] = o; bo[j] = f2bf(o);
  }
}

// ---------------- kernel 5: generic bf16 WMMA GEMM, C = A[M,K] * W[N,K]^T + bias ----------------
// block = 256 threads = 8 waves; wave -> 16 x (16*NW) tile (A fragment reused NW times)
// grid = (ceil(N/(128*NW)), M/16)
template <int NW>
__global__ void k_gemm_nt(const u16* __restrict__ A, const u16* __restrict__ W,
                          const float* __restrict__ bias, float* __restrict__ Cf,
                          u16* __restrict__ Cb, int M, int N, int K, int relu) {
  const int wave = threadIdx.x >> 5;
  const int lane = threadIdx.x & 31;
  const int n0 = (blockIdx.x * 8 + wave) * (16 * NW);
  const int m0 = blockIdx.y * 16;
  if (n0 >= N) return;
  const int half = lane >> 4;
  const int r16  = lane & 15;
  const u16* arow = A + (size_t)(m0 + r16) * K;
  const u16* wrow[NW];
  v8f acc[NW];
#pragma unroll
  for (int w = 0; w < NW; ++w) {
    wrow[w] = W + (size_t)(n0 + w * 16 + r16) * K;
    acc[w] = (v8f){0.f,0.f,0.f,0.f,0.f,0.f,0.f,0.f};
  }
  const bool deep = (K > 256);
  for (int k0 = 0; k0 < K; k0 += 32) {
    if (deep && k0 + 256 < K) {     // gfx1250 speculative prefetch of B rows 8 k-steps ahead
      __builtin_prefetch(wrow[0] + k0 + 256, 0, 1);
      __builtin_prefetch(arow + k0 + 256, 0, 1);
    }
    v16bf af = load_a(arow, k0, half);
#pragma unroll
    for (int w = 0; w < NW; ++w) {
      acc[w] = __builtin_amdgcn_wmma_f32_16x16x32_bf16(false, af,
                                                       false, load_b(wrow[w], k0, half),
                                                       (short)0, acc[w], false, false);
    }
  }
#pragma unroll
  for (int w = 0; w < NW; ++w) {
    const int n = n0 + w * 16 + r16;
    const float bn = bias ? bias[n] : 0.f;
#pragma unroll
    for (int r = 0; r < 8; ++r) {
      const int m = m0 + half * 8 + r;
      float v = acc[w][r] + bn;
      if (relu) v = fmaxf(v, 0.f);
      const size_t o = (size_t)m * N + n;
      if (Cf) Cf[o] = v;
      if (Cb) Cb[o] = f2bf(v);
    }
  }
}

// ---------------- kernel 6: attention, one block(64 thr) per (b,h), LDS-staged ----------------
__global__ void k_attn(const float* __restrict__ qkv, const int* __restrict__ seq,
                       u16* __restrict__ ob) {
  __shared__ float q_s[SS*DHH], k_s[SS*DHH], v_s[SS*DHH], p_s[SS*SS];
  const int b = blockIdx.x >> 1;
  const int h = blockIdx.x & 1;
  const int tid = threadIdx.x;
  for (int i = tid; i < SS * DHH; i += 64) {
    int s = i / DHH, d = i % DHH;
    const float* base = qkv + ((size_t)b * SS + s) * (3 * DD) + h * DHH + d;
    q_s[i] = base[0];
    k_s[i] = base[DD];
    v_s[i] = base[2 * DD];
  }
  __syncthreads();
  if (tid < SS) {
    const int qi = tid;
    float mx = -1e30f;
    for (int kj = 0; kj < SS; ++kj) {
      float dot = 0.f;
#pragma unroll 8
      for (int d = 0; d < DHH; ++d) dot += q_s[qi*DHH+d] * k_s[kj*DHH+d];
      dot = dot * 0.125f + (seq[b*SS+kj] == 0 ? -1e9f : 0.f);
      p_s[qi*SS+kj] = dot;
      mx = fmaxf(mx, dot);
    }
    float se = 0.f;
    for (int kj = 0; kj < SS; ++kj) { float e = __expf(p_s[qi*SS+kj] - mx); p_s[qi*SS+kj] = e; se += e; }
    float inv = 1.f / se;
    u16* orow = ob + ((size_t)b * SS + qi) * DD + h * DHH;
    for (int d = 0; d < DHH; ++d) {
      float o = 0.f;
#pragma unroll 10
      for (int kj = 0; kj < SS; ++kj) o += p_s[qi*SS+kj] * v_s[kj*DHH+d];
      orow[d] = f2bf(o * inv);
    }
  }
}

// ---------------- kernel 7: cond = l2norm(x[:, -1, :]) ----------------
__global__ void k_cond(const float* __restrict__ x, float* __restrict__ cond,
                       u16* __restrict__ condb) {
  int b = blockIdx.x * 8 + (threadIdx.x >> 5);
  if (b >= BB) return;
  int lane = threadIdx.x & 31;
  float4 v = *(const float4*)(x + ((size_t)b * SS + SS - 1) * DD + lane * 4);
  float ss = wsum(v.x*v.x + v.y*v.y + v.z*v.z + v.w*v.w);
  float inv = 1.0f / fmaxf(sqrtf(ss), 1e-12f);
  float* co = cond + (size_t)b * DD + lane * 4;
  u16*   cb = condb + (size_t)b * DD + lane * 4;
  co[0]=v.x*inv; co[1]=v.y*inv; co[2]=v.z*inv; co[3]=v.w*inv;
  cb[0]=f2bf(v.x*inv); cb[1]=f2bf(v.y*inv); cb[2]=f2bf(v.z*inv); cb[3]=f2bf(v.w*inv);
}

// ---------------- kernel 8: t-embedding (t==1 for all b -> single 128-vec) ----------------
__global__ void k_temb(const float* __restrict__ tw1, const float* __restrict__ tb1,
                       const float* __restrict__ tw2, const float* __restrict__ tb2,
                       float* __restrict__ temb) {
  __shared__ float a[DD];
  int t = threadIdx.x;
  a[t] = silu_f(tw1[t] + tb1[t]);
  __syncthreads();
  float s = tb2[t];
  for (int j = 0; j < DD; ++j) s += tw2[t*DD+j] * a[j];
  temb[t] = s;
}

// ---------------- kernel 9: GMFlow head, one block(128 thr) per batch row ----------------
__global__ void k_head(const float* __restrict__ xT, const float* __restrict__ cond,
                       const float* __restrict__ temb,
                       const float* __restrict__ fw, const float* __restrict__ fb,
                       const float* __restrict__ bw1, const float* __restrict__ bb1,
                       const float* __restrict__ bw2, const float* __restrict__ bb2,
                       const float* __restrict__ mw, const float* __restrict__ mb,
                       const float* __restrict__ ww, const float* __restrict__ wb,
                       const float* __restrict__ sw1, const float* __restrict__ sb1,
                       const float* __restrict__ sw2, const float* __restrict__ sb2,
                       u16* __restrict__ meansb, float* __restrict__ ckb,
                       float* __restrict__ scale) {
  __shared__ float in0[2*DD], h[DD], h2[DD], ms[KK][DD], red[DD], wts[KK], u[DHH];
  __shared__ float sc_s;
  const int b = blockIdx.x;
  const int t = threadIdx.x;
  in0[t]      = xT[b*DD + t];
  in0[DD + t] = cond[b*DD + t];
  __syncthreads();
  // h = silu([xT,cond] @ fw.T + fb) + t_emb
  float a = fb[t];
  for (int j = 0; j < 2*DD; ++j) a += fw[t*2*DD + j] * in0[j];
  h[t] = silu_f(a) + temb[t];
  __syncthreads();
  a = bb1[t];
  for (int j = 0; j < DD; ++j) a += bw1[t*DD + j] * h[j];
  h2[t] = silu_f(a);
  __syncthreads();
  a = bb2[t];
  for (int j = 0; j < DD; ++j) a += bw2[t*DD + j] * h2[j];
  h[t] = silu_f(a);           // final hidden
  __syncthreads();
  // means (pre-norm)
  for (int k = 0; k < KK; ++k) {
    float mk = mb[k*DD + t];
    for (int j = 0; j < DD; ++j) mk += mw[(size_t)(k*DD + t)*DD + j] * h[j];
    ms[k][t] = mk;
  }
  // mixture logits + scale branch
  if (t < KK) {
    float wl = wb[t];
    for (int j = 0; j < DD; ++j) wl += ww[t*DD + j] * h[j];
    wts[t] = wl;
  }
  if (t < DHH) {
    float su = sb1[t];
    for (int j = 0; j < DD; ++j) su += sw1[t*(DD+1) + j] * cond[b*DD + j];
    su += sw1[t*(DD+1) + DD];       // t == 1
    u[t] = silu_f(su);
  }
  __syncthreads();
  if (t == 0) {
    float ls = sb2[0];
    for (int j = 0; j < DHH; ++j) ls += sw2[j] * u[j];
    float s = log1pf(__expf(ls)) + 0.05f;
    sc_s = 1.0f / (s * s);
    scale[b] = sc_s;
    float mx = fmaxf(fmaxf(wts[0], wts[1]), fmaxf(wts[2], wts[3]));
    float e0 = __expf(wts[0]-mx), e1 = __expf(wts[1]-mx),
          e2 = __expf(wts[2]-mx), e3 = __expf(wts[3]-mx);
    float se = e0 + e1 + e2 + e3;
    wts[0] = e0/se; wts[1] = e1/se; wts[2] = e2/se; wts[3] = e3/se;
  }
  __syncthreads();
  // l2norm each mean row, emit bf16 k-major matrix + per-(b,k) logsumexp constant
  for (int k = 0; k < KK; ++k) {
    red[t] = ms[k][t] * ms[k][t];
    __syncthreads();
    for (int s2 = 64; s2 > 0; s2 >>= 1) {
      if (t < s2) red[t] += red[t + s2];
      __syncthreads();
    }
    float ss = red[0];
    float d = fmaxf(sqrtf(ss), 1e-12f);
    float inv = 1.0f / d;
    meansb[((size_t)k * BB + b) * DD + t] = f2bf(ms[k][t] * inv);
    if (t == 0) {
      float nq = ss * inv * inv;
      ckb[b*KK + k] = __logf(wts[k] + 1e-9f) - 0.5f * nq * sc_s;
    }
    __syncthreads();
  }
}

// ---------------- scoring epilogue: 4-way logsumexp + both score planes ----------------
__device__ __forceinline__ void score_epilogue(const v8f& accc, const v8f* accm,
                                               int b0, int half, int n,
                                               const float* __restrict__ nsq,
                                               const float* __restrict__ ckb,
                                               const float* __restrict__ scale,
                                               float* __restrict__ sa,
                                               float* __restrict__ lg) {
  const float nq = nsq[n];
#pragma unroll
  for (int r = 0; r < 8; ++r) {
    const int b = b0 + half * 8 + r;
    const float sc = scale[b];
    float t0 = ckb[b*KK+0] + accm[0][r] * sc;
    float t1 = ckb[b*KK+1] + accm[1][r] * sc;
    float t2 = ckb[b*KK+2] + accm[2][r] * sc;
    float t3 = ckb[b*KK+3] + accm[3][r] * sc;
    float mx = fmaxf(fmaxf(t0, t1), fmaxf(t2, t3));
    float se = __expf(t0-mx) + __expf(t1-mx) + __expf(t2-mx) + __expf(t3-mx);
    const size_t o = (size_t)b * NN + n;
    sa[o] = accc[r];
    lg[o] = mx + __logf(se) - 0.5f * nq * sc;
  }
}

// ---------------- kernel 10: fused catalog scoring (cond-dot + 4-way GM logsumexp) ----------------
// wave -> 16(batch) x 32(item) tile: 5 A fragments (cond + 4 means) reused across 2 item
// B fragments -> 10 WMMAs per k-step for 7 fragment loads. Branchless tail: the second
// subtile's row pointer is clamped (redundant compute, guarded stores) so the k-loop has
// no divergent control flow and no exec-mask churn -> no VGPR spills.
__global__ void k_score(const u16* __restrict__ condb, const u16* __restrict__ meansb,
                        const u16* __restrict__ items, const float* __restrict__ nsq,
                        const float* __restrict__ ckb, const float* __restrict__ scale,
                        float* __restrict__ sa, float* __restrict__ lg) {
  const int wave = threadIdx.x >> 5;
  const int lane = threadIdx.x & 31;
  const int n0 = (blockIdx.x * 8 + wave) * 32;
  if (n0 >= NN) return;
  const bool tile1 = (n0 + 16) < NN;          // NN % 16 == 0 -> all-or-nothing per subtile
  const int n1 = tile1 ? (n0 + 16) : n0;      // clamp on tail (redundant compute, no OOB)
  const int b0 = blockIdx.y * 16;
  const int half = lane >> 4;
  const int r16  = lane & 15;
  const u16* crow  = condb + (size_t)(b0 + r16) * DD;
  const u16* irow0 = items + (size_t)(n0 + r16) * DD;
  const u16* irow1 = items + (size_t)(n1 + r16) * DD;
  const u16* mrow[KK];
#pragma unroll
  for (int k = 0; k < KK; ++k) mrow[k] = meansb + ((size_t)k * BB + b0 + r16) * DD;
  v8f accc0 = (v8f){0.f,0.f,0.f,0.f,0.f,0.f,0.f,0.f};
  v8f accc1 = accc0;
  v8f accm0[KK], accm1[KK];
#pragma unroll
  for (int k = 0; k < KK; ++k) { accm0[k] = accc0; accm1[k] = accc0; }
#pragma unroll 1
  for (int k0 = 0; k0 < DD; k0 += 32) {
    v16bf bf0 = load_b(irow0, k0, half);
    v16bf bf1 = load_b(irow1, k0, half);
    v16bf af  = load_a(crow, k0, half);
    accc0 = __builtin_amdgcn_wmma_f32_16x16x32_bf16(false, af, false, bf0,
                                                    (short)0, accc0, false, false);
    accc1 = __builtin_amdgcn_wmma_f32_16x16x32_bf16(false, af, false, bf1,
                                                    (short)0, accc1, false, false);
#pragma unroll
    for (int k = 0; k < KK; ++k) {
      v16bf am = load_a(mrow[k], k0, half);
      accm0[k] = __builtin_amdgcn_wmma_f32_16x16x32_bf16(false, am, false, bf0,
                                                         (short)0, accm0[k], false, false);
      accm1[k] = __builtin_amdgcn_wmma_f32_16x16x32_bf16(false, am, false, bf1,
                                                         (short)0, accm1[k], false, false);
    }
  }
  score_epilogue(accc0, accm0, b0, half, n0 + r16, nsq, ckb, scale, sa, lg);
  if (tile1)
    score_epilogue(accc1, accm1, b0, half, n1 + r16, nsq, ckb, scale, sa, lg);
}

// ---------------- kernel 11: per-row mean / unbiased std ----------------
__global__ void k_stats(const float* __restrict__ sa, const float* __restrict__ lg,
                        float* __restrict__ stats) {
  __shared__ float r0[256], r1[256], r2[256], r3[256];
  const int b = blockIdx.x;
  const int t = threadIdx.x;
  float s0 = 0.f, q0 = 0.f, s1 = 0.f, q1 = 0.f;
  for (int n = t; n < NN; n += 256) {
    float x = sa[(size_t)b * NN + n]; s0 += x; q0 += x * x;
    float y = lg[(size_t)b * NN + n]; s1 += y; q1 += y * y;
  }
  r0[t] = s0; r1[t] = q0; r2[t] = s1; r3[t] = q1;
  __syncthreads();
  for (int s2 = 128; s2 > 0; s2 >>= 1) {
    if (t < s2) { r0[t]+=r0[t+s2]; r1[t]+=r1[t+s2]; r2[t]+=r2[t+s2]; r3[t]+=r3[t+s2]; }
    __syncthreads();
  }
  if (t == 0) {
    float m0 = r0[0] / NN;
    float v0 = (r1[0] - (float)NN * m0 * m0) / (float)(NN - 1);
    float m1 = r2[0] / NN;
    float v1 = (r3[0] - (float)NN * m1 * m1) / (float)(NN - 1);
    stats[b*4+0] = m0; stats[b*4+1] = 1.0f / (sqrtf(fmaxf(v0, 0.f)) + 1e-6f);
    stats[b*4+2] = m1; stats[b*4+3] = 1.0f / (sqrtf(fmaxf(v1, 0.f)) + 1e-6f);
  }
}

// ---------------- kernel 12: score fusion ----------------
__global__ void k_fuse(const float* __restrict__ sa, const float* __restrict__ lg,
                       const float* __restrict__ stats, float* __restrict__ out) {
  size_t i = (size_t)blockIdx.x * 256 + threadIdx.x;
  if (i >= (size_t)BB * NN) return;
  int b = (int)(i / NN);
  int n = (int)(i % NN);
  const float* st = stats + b * 4;
  float v = 0.5f * (sa[i] - st[0]) * st[1] + 0.5f * (lg[i] - st[2]) * st[3];
  out[i] = (n == 0) ? -__builtin_inff() : v;
}

// ---------------- host orchestration ----------------
extern "C" void kernel_launch(void* const* d_in, const int* in_sizes, int n_in,
                              void* d_out, int out_size, void* d_ws, size_t ws_size,
                              hipStream_t stream) {
  (void)in_sizes; (void)n_in; (void)out_size; (void)ws_size;
  const float* item_emb  = (const float*)d_in[0];
  const float* pos_emb   = (const float*)d_in[1];
  const float* ln0_w     = (const float*)d_in[2];
  const float* ln0_b     = (const float*)d_in[3];
  const float* enc_wqkv  = (const float*)d_in[4];
  const float* enc_bqkv  = (const float*)d_in[5];
  const float* enc_wo    = (const float*)d_in[6];
  const float* enc_bo    = (const float*)d_in[7];
  const float* enc_ln1_w = (const float*)d_in[8];
  const float* enc_ln1_b = (const float*)d_in[9];
  const float* enc_w1    = (const float*)d_in[10];
  const float* enc_b1    = (const float*)d_in[11];
  const float* enc_w2    = (const float*)d_in[12];
  const float* enc_b2    = (const float*)d_in[13];
  const float* enc_ln2_w = (const float*)d_in[14];
  const float* enc_ln2_b = (const float*)d_in[15];
  const float* tw1 = (const float*)d_in[16];
  const float* tb1 = (const float*)d_in[17];
  const float* tw2 = (const float*)d_in[18];
  const float* tb2 = (const float*)d_in[19];
  const float* fw  = (const float*)d_in[20];
  const float* fb  = (const float*)d_in[21];
  const float* bw1 = (const float*)d_in[22];
  const float* bb1 = (const float*)d_in[23];
  const float* bw2 = (const float*)d_in[24];
  const float* bb2 = (const float*)d_in[25];
  const float* mw  = (const float*)d_in[26];
  const float* mb  = (const float*)d_in[27];
  const float* ww  = (const float*)d_in[28];
  const float* wb  = (const float*)d_in[29];
  const float* sw1 = (const float*)d_in[30];
  const float* sb1 = (const float*)d_in[31];
  const float* sw2 = (const float*)d_in[32];
  const float* sb2 = (const float*)d_in[33];
  const float* xT  = (const float*)d_in[34];
  const int* item_seq = (const int*)d_in[35];
  float* out = (float*)d_out;

  char* ws = (char*)d_ws;
  size_t off = 0;
  auto alloc = [&](size_t bytes) { size_t r = off; off = (off + bytes + 255) & ~(size_t)255; return r; };

  // persistent region (~16 MB)
  u16*   items_bf = (u16*)  (ws + alloc((size_t)NN * DD * 2));
  float* item_nsq = (float*)(ws + alloc((size_t)NN * 4));
  u16*   condb    = (u16*)  (ws + alloc((size_t)BB * DD * 2));
  float* cond     = (float*)(ws + alloc((size_t)BB * DD * 4));
  u16*   meansb   = (u16*)  (ws + alloc((size_t)KK * BB * DD * 2));
  float* ckb      = (float*)(ws + alloc((size_t)BB * KK * 4));
  float* scalep   = (float*)(ws + alloc((size_t)BB * 4));
  float* temb     = (float*)(ws + alloc((size_t)DD * 4));
  float* stats    = (float*)(ws + alloc((size_t)BB * 4 * 4));
  u16* wqkv_bf = (u16*)(ws + alloc((size_t)NLL * 3 * DD * DD * 2));
  u16* wo_bf   = (u16*)(ws + alloc((size_t)NLL * DD * DD * 2));
  u16* w1_bf   = (u16*)(ws + alloc((size_t)NLL * DFFF * DD * 2));
  u16* w2_bf   = (u16*)(ws + alloc((size_t)NLL * DD * DFFF * 2));

  // temp region: encoder scratch, later aliased by the two score planes
  const size_t tbase = off;
  size_t toff = tbase;
  auto talloc = [&](size_t bytes) { size_t r = toff; toff = (toff + bytes + 255) & ~(size_t)255; return r; };
  float* x    = (float*)(ws + talloc((size_t)BB * SS * DD * 4));
  u16*   xb   = (u16*)  (ws + talloc((size_t)BB * SS * DD * 2));
  float* qkv  = (float*)(ws + talloc((size_t)BB * SS * 3 * DD * 4));
  u16*   ob   = (u16*)  (ws + talloc((size_t)BB * SS * DD * 2));
  float* proj = (float*)(ws + talloc((size_t)BB * SS * DD * 4));
  u16*   h1   = (u16*)  (ws + talloc((size_t)BB * SS * DFFF * 2));
  // scoring view (aliases encoder scratch; encoder results consumed before these writes)
  size_t soff = tbase;
  float* sa = (float*)(ws + soff); soff = (soff + (size_t)BB * NN * 4 + 255) & ~(size_t)255;
  float* lg = (float*)(ws + soff);

  const int MROWS = BB * SS;   // 12800

  // 1) item table prep + weight conversion
  k_items<<<(NN + 7) / 8, 256, 0, stream>>>(item_emb, items_bf, item_nsq);
  k_cvt<<<(NLL*3*DD*DD + 255) / 256, 256, 0, stream>>>(enc_wqkv, wqkv_bf, NLL*3*DD*DD);
  k_cvt<<<(NLL*DD*DD   + 255) / 256, 256, 0, stream>>>(enc_wo,   wo_bf,   NLL*DD*DD);
  k_cvt<<<(NLL*DFFF*DD + 255) / 256, 256, 0, stream>>>(enc_w1,   w1_bf,   NLL*DFFF*DD);
  k_cvt<<<(NLL*DD*DFFF + 255) / 256, 256, 0, stream>>>(enc_w2,   w2_bf,   NLL*DD*DFFF);

  // 2) embedding + LN0
  k_embed<<<(MROWS + 7) / 8, 256, 0, stream>>>(item_emb, pos_emb, item_seq, ln0_w, ln0_b, x, xb);

  // 3) transformer encoder layers
  for (int i = 0; i < NLL; ++i) {
    // QKV: N=384, NW=2 -> 12 wave-tiles, grid.x = 2
    k_gemm_nt<2><<<dim3(2, MROWS/16), 256, 0, stream>>>(
        xb, wqkv_bf + (size_t)i*3*DD*DD, enc_bqkv + i*3*DD, qkv, (u16*)nullptr,
        MROWS, 3*DD, DD, 0);
    k_attn<<<BB * HH, 64, 0, stream>>>(qkv, item_seq, ob);
    // O-proj: N=128, NW=1 -> 8 wave-tiles, grid.x = 1
    k_gemm_nt<1><<<dim3(1, MROWS/16), 256, 0, stream>>>(
        ob, wo_bf + (size_t)i*DD*DD, enc_bo + i*DD, proj, (u16*)nullptr,
        MROWS, DD, DD, 0);
    k_resln<<<(MROWS + 7) / 8, 256, 0, stream>>>(x, proj, enc_ln1_w + i*DD, enc_ln1_b + i*DD, xb);
    // FFN1: N=2048, NW=4 -> A fragment reused 4x, grid.x = 2048/(64*8) = 4
    k_gemm_nt<4><<<dim3(4, MROWS/16), 256, 0, stream>>>(
        xb, w1_bf + (size_t)i*DFFF*DD, enc_b1 + i*DFFF, (float*)nullptr, h1,
        MROWS, DFFF, DD, 1);
    // FFN2: N=128, K=2048 (deep-K path prefetches), NW=1
    k_gemm_nt<1><<<dim3(1, MROWS/16), 256, 0, stream>>>(
        h1, w2_bf + (size_t)i*DD*DFFF, enc_b2 + i*DD, proj, (u16*)nullptr,
        MROWS, DD, DFFF, 0);
    k_resln<<<(MROWS + 7) / 8, 256, 0, stream>>>(x, proj, enc_ln2_w + i*DD, enc_ln2_b + i*DD, xb);
  }

  // 4) cond + GMFlow head
  k_cond<<<(BB + 7) / 8, 256, 0, stream>>>(x, cond, condb);
  k_temb<<<1, DD, 0, stream>>>(tw1, tb1, tw2, tb2, temb);
  k_head<<<BB, DD, 0, stream>>>(xT, cond, temb, fw, fb, bw1, bb1, bw2, bb2,
                                mw, mb, ww, wb, sw1, sb1, sw2, sb2,
                                meansb, ckb, scalep);

  // 5) fused catalog scoring (overwrites encoder scratch region)
  //    wave tile = 16 batches x 32 items; tiles32 = ceil(50000/32) = 1563; grid.x = 196
  k_score<<<dim3(((NN + 31) / 32 + 7) / 8, BB/16), 256, 0, stream>>>(
      condb, meansb, items_bf, item_nsq, ckb, scalep, sa, lg);

  // 6) row stats + fusion
  k_stats<<<BB, 256, 0, stream>>>(sa, lg, stats);
  k_fuse<<<(int)(((size_t)BB * NN + 255) / 256), 256, 0, stream>>>(sa, lg, stats, out);
}
